// DeterministicActorRNN_31044023615743
// MI455X (gfx1250) — matrix operand order repaired and checked by reading, not compiled, not run on hardware
//
#include <hip/hip_runtime.h>
#include <hip/hip_bf16.h>

// ---------------- types ----------------
typedef _Float16 f16_t;
typedef __attribute__((ext_vector_type(16))) _Float16 v16h;
typedef __attribute__((ext_vector_type(8)))  _Float16 v8h;
typedef __attribute__((ext_vector_type(8)))  float    v8f;

#define BATCH    16384
#define HDIM     256
#define GOAL     64
#define TSTEPS   5
#define MLP_IN   320     // HDIM + GOAL
#define MLP_HID  1024
#define MLP_OUT  64
#define STATE_LD 1344    // OBS_DIM + GOAL_DIM
#define MT       64      // batch tile rows per workgroup

// LDS leading dims (halves); row strides = odd multiples of 16B to stagger banks
#define LDH 264          // RNN hidden tiles  (256 + 8 pad)   row = 528B
#define LDM 1032         // MLP hidden tile   (1024 + 8 pad)  row = 2064B
#define LDC 72           // MLP h2 chunk tile (64 + 8 pad)    row = 144B
#define SWLD 40          // staged weight slice row stride    row = 80B (odd*16B)
#define SW_MAT (HDIM * SWLD)   // halves per staged 256x32 matrix slice

// ---------------- workspace layout (bytes) ----------------
#define OFF_WIH   ((size_t)0)                                    // 2*256*256 f16
#define OFF_WHH   (OFF_WIH  + (size_t)2*HDIM*HDIM*2)
#define OFF_W1    (OFF_WHH  + (size_t)2*HDIM*HDIM*2)             // 1024*320 f16
#define OFF_W2    (OFF_W1   + (size_t)MLP_HID*MLP_IN*2)          // 1024*1024 f16
#define OFF_W3    (OFF_W2   + (size_t)MLP_HID*MLP_HID*2)         // 64*1024 f16
#define OFF_BSUM  (OFF_W3   + (size_t)MLP_OUT*MLP_HID*2)         // 2*256 f32
#define OFF_XSEQ  (OFF_BSUM + (size_t)2*HDIM*4 + 2048)           // 5*B*256 f16 (time-reversed)
#define OFF_COMB  (OFF_XSEQ + (size_t)TSTEPS*BATCH*HDIM*2)       // B*320 f16

// ---------------- CDNA5 async global->LDS copy (ASYNCcnt path) ----------------
// Copies 16B per active lane from global to LDS. LDS operand is the low 32 bits
// of the flat pointer (ISA 10.2: LDS aperture maps addr[31:0] directly).
__device__ __forceinline__ void async_cp_b128(const f16_t* g, f16_t* l) {
  unsigned lofs = (unsigned)(uintptr_t)l;
  unsigned long long ga = (unsigned long long)(uintptr_t)g;
  asm volatile("global_load_async_to_lds_b128 %0, %1, off"
               :: "v"(lofs), "v"(ga) : "memory");
}
__device__ __forceinline__ void wait_async0() {
  asm volatile("s_wait_asynccnt 0" ::: "memory");
}

// Stage k-slice kb (32 cols) of two 256x256 f16 matrices into dst (LDS):
// dst[0..SW_MAT) = Wa[:, kb*32..), dst[SW_MAT..) = Wb[:, kb*32..).
// 1024 16B-chunks per matrix; 256 threads -> 4 chunks each -> 8 asyncs/thread.
__device__ __forceinline__ void stage_w_slice(f16_t* dst, const f16_t* Wa, const f16_t* Wb,
                                              int kb, int tid) {
#pragma unroll
  for (int i = 0; i < 4; ++i) {
    const int c   = tid + i * 256;     // chunk id 0..1023
    const int row = c >> 2;            // weight row 0..255
    const int cc  = (c & 3) * 8;       // 8-half chunk offset in row
    async_cp_b128(Wa + (size_t)row * HDIM + kb * 32 + cc, dst + row * SWLD + cc);
    async_cp_b128(Wb + (size_t)row * HDIM + kb * 32 + cc, dst + SW_MAT + row * SWLD + cc);
  }
}

// ---------------- WMMA fragment helpers (contiguous k = 16h + s in both A and B) ----------------
// A (16x32 f16, M x K): lane L<16 holds row L, K 0..15 contiguous; lane 16+L -> K 16..31
__device__ __forceinline__ v16h load_a(const f16_t* X, int ld, int row0, int k0, int lane) {
  const int r = row0 + (lane & 15);
  const int c = k0 + (lane & 16);                  // +16 for upper half-wave
  const f16_t* p = X + (size_t)r * ld + c;
  v8h lo = *(const v8h*)p;
  v8h hi = *(const v8h*)(p + 8);
  v16h out;
#pragma unroll
  for (int i = 0; i < 8; ++i) { out[i] = lo[i]; out[i + 8] = hi[i]; }
  return out;
}

// B (32x16 f16, K x N) for out = X * W^T, W stored row-major [N][K]:
// lane n<16 holds col n, K 0..15 contiguous; lane 16+n holds K 16..31.
__device__ __forceinline__ v16h load_b(const f16_t* W, int ld, int n0, int k0, int lane) {
  const int n = n0 + (lane & 15);
  const int k = k0 + (lane & 16);                  // +16 for upper half-wave
  return *(const v16h*)(W + (size_t)n * ld + k);   // 32 contiguous bytes
}

__device__ __forceinline__ v8f wmma_f16(v16h a, v16h b, v8f c) {
  return __builtin_amdgcn_wmma_f32_16x16x32_f16(false, a, false, b, (short)0, c, false, false);
}

// C/D (16x16 f32): VGPR r, lane n<16 -> (M=r, N=n); lane 16+n -> (M=8+r, N=n)

// ---------------- phase 0: conversions ----------------
__global__ void k_cvt_f16(const float* __restrict__ src, f16_t* __restrict__ dst, int n) {
  int i = blockIdx.x * blockDim.x + threadIdx.x;
  if (i < n) dst[i] = (f16_t)src[i];
}

__global__ void k_bias_sum(const float* __restrict__ bih, const float* __restrict__ bhh,
                           float* __restrict__ bsum, int n) {
  int i = blockIdx.x * blockDim.x + threadIdx.x;
  if (i < n) bsum[i] = bih[i] + bhh[i];
}

// time-reverse obs frames into xseq[t][b][256] and drop goal into comb[b][256..320)
__global__ void k_prep(const float* __restrict__ state, f16_t* __restrict__ xseq,
                       f16_t* __restrict__ comb) {
  int i = blockIdx.x * blockDim.x + threadIdx.x;
  if (i >= BATCH * HDIM) return;
  int b = i >> 8;
  int j = i & 255;
  const float* srow = state + (size_t)b * STATE_LD;
#pragma unroll
  for (int t = 0; t < TSTEPS; ++t)
    xseq[((size_t)t * BATCH + b) * HDIM + j] = (f16_t)srow[(TSTEPS - 1 - t) * HDIM + j];
  if (j < GOAL)
    comb[(size_t)b * MLP_IN + HDIM + j] = (f16_t)srow[TSTEPS * HDIM + j];
}

// ---------------- RNN: 2-layer tanh RNN, batch tile = 64 rows ----------------
// 8 waves: wm = wave&3 (4 M-tiles), wn = wave>>2 (2 N-groups of 128 cols)
// Weight k-slices are async-staged into LDS, double-buffered, overlapping WMMAs.
__global__ __launch_bounds__(256) void k_rnn(const f16_t* __restrict__ xseq,
                                             const f16_t* __restrict__ Wih,
                                             const f16_t* __restrict__ Whh,
                                             const float* __restrict__ bsum,
                                             f16_t* __restrict__ comb) {
  __shared__ __align__(32) f16_t sH0[MT * LDH];
  __shared__ __align__(32) f16_t sH1[MT * LDH];
  __shared__ __align__(32) f16_t sW[2][2 * SW_MAT];   // [buf][Wi slice | Wh slice]

  const int tid  = threadIdx.x;
  const int lane = tid & 31;
  const int wave = tid >> 5;
  const int row0 = (wave & 3) * 16;
  const int nbas = (wave >> 2) * 128;
  const size_t wgRow = (size_t)blockIdx.x * MT;

  for (int i = tid; i < MT * LDH; i += 256) { sH0[i] = (f16_t)0; sH1[i] = (f16_t)0; }
  __syncthreads();

  for (int t = 0; t < TSTEPS; ++t) {
    const f16_t* X = xseq + ((size_t)t * BATCH + wgRow) * HDIM;
    if (t + 1 < TSTEPS) {   // global_prefetch of next timestep's input tile (128B/lane)
      const f16_t* Xn = X + (size_t)BATCH * HDIM;
      __builtin_prefetch((const void*)(Xn + (size_t)tid * 64), 0, 0);
    }

#pragma unroll
    for (int layer = 0; layer < 2; ++layer) {
      const f16_t* Wi = Wih + (size_t)layer * HDIM * HDIM;
      const f16_t* Wh = Whh + (size_t)layer * HDIM * HDIM;
      const f16_t* A0 = (layer == 0) ? X   : sH0;   // input-activation source
      const int    a0ld = (layer == 0) ? HDIM : LDH;
      f16_t*       Hst  = (layer == 0) ? sH0 : sH1; // state being updated
      const float* bs   = bsum + layer * HDIM;

      // prologue: stage k-slice 0
      stage_w_slice(sW[0], Wi, Wh, 0, tid);
      wait_async0();
      __syncthreads();

      v8f acc[8] = {};
      for (int kb = 0; kb < 8; ++kb) {            // K = 8 x 32
        if (kb + 1 < 8) stage_w_slice(sW[(kb + 1) & 1], Wi, Wh, kb + 1, tid);
        const f16_t* sw = sW[kb & 1];
        const int k0 = kb * 32;
        v16h ax = load_a(A0,  a0ld, row0, k0, lane);
        v16h ah = load_a(Hst, LDH,  row0, k0, lane);
#pragma unroll
        for (int j = 0; j < 8; ++j) {
          const int n0 = nbas + j * 16;
          acc[j] = wmma_f16(ax, load_b(sw,          SWLD, n0, 0, lane), acc[j]);
          acc[j] = wmma_f16(ah, load_b(sw + SW_MAT, SWLD, n0, 0, lane), acc[j]);
        }
        if (kb + 1 < 8) { wait_async0(); __syncthreads(); }
      }
      __syncthreads();                            // all reads of Hst complete
#pragma unroll
      for (int j = 0; j < 8; ++j) {
        const int n  = nbas + j * 16 + (lane & 15);
        const float bv = bs[n];
        const int rb = row0 + ((lane & 16) >> 1);
#pragma unroll
        for (int r = 0; r < 8; ++r)
          Hst[(rb + r) * LDH + n] = (f16_t)tanhf(acc[j][r] + bv);
      }
      __syncthreads();
    }
  }

  // final hidden -> comb[:, 0:256]
  for (int i = tid; i < MT * HDIM; i += 256) {
    int r = i >> 8, c = i & 255;
    comb[(wgRow + r) * MLP_IN + c] = sH1[r * LDH + c];
  }
}

// ---------------- MLP: 320 -> 1024 -> 1024 -> 64, batch tile = 64 ----------------
// Layer1: A-frags straight from global comb -> sA (relu).
// Layer2+3 fused: h2 produced in 64-col chunks into sChunk, immediately consumed
// as the K-chunk of layer3 whose accumulators persist in VGPRs.
__global__ __launch_bounds__(256) void k_mlp(const f16_t* __restrict__ comb,
                                             const f16_t* __restrict__ W1, const float* __restrict__ b1,
                                             const f16_t* __restrict__ W2, const float* __restrict__ b2,
                                             const f16_t* __restrict__ W3, const float* __restrict__ b3,
                                             float* __restrict__ out) {
  __shared__ __align__(32) f16_t sA[MT * LDM];       // 64 x 1024 h1 (relu)
  __shared__ __align__(32) f16_t sChunk[MT * LDC];   // 64 x 64 h2 chunk (relu)

  const int tid  = threadIdx.x;
  const int lane = tid & 31;
  const int wave = tid >> 5;
  const int wm   = wave & 3;          // M-tile
  const int wn   = wave >> 2;         // N-half
  const int row0 = wm * 16;
  const size_t wgRow = (size_t)blockIdx.x * MT;
  const f16_t* inG = comb + wgRow * MLP_IN;          // global A for layer 1

  // ---- layer 1: relu(in @ W1^T + b1) -> sA ; N=1024, K=320 ----
  for (int cch = 0; cch < 8; ++cch) {
    const int nb = wn * 512 + cch * 64;
    v8f acc[4] = {};
    for (int k0 = 0; k0 < MLP_IN; k0 += 32) {
      v16h a = load_a(inG, MLP_IN, row0, k0, lane);
#pragma unroll
      for (int j = 0; j < 4; ++j)
        acc[j] = wmma_f16(a, load_b(W1, MLP_IN, nb + j * 16, k0, lane), acc[j]);
    }
#pragma unroll
    for (int j = 0; j < 4; ++j) {
      const int n  = nb + j * 16 + (lane & 15);
      const float bv = b1[n];
      const int rb = row0 + ((lane & 16) >> 1);
#pragma unroll
      for (int r = 0; r < 8; ++r)
        sA[(rb + r) * LDM + n] = (f16_t)fmaxf(acc[j][r] + bv, 0.0f);
    }
  }
  __syncthreads();

  // ---- layers 2+3 fused over 16 chunks of 64 h2-columns ----
  v8f acc3[2] = {};
  for (int kc = 0; kc < 16; ++kc) {
    // layer 2 partial: h2[:, kc*64 .. +64) -> sChunk
    {
      if (kc + 1 < 16)  // prefetch next chunk's W2 rows
        __builtin_prefetch((const void*)(W2 + ((size_t)(kc + 1) * 64 + (tid & 63)) * MLP_HID), 0, 0);
      v8f acc[2] = {};
      for (int k0 = 0; k0 < MLP_HID; k0 += 32) {
        v16h a = load_a(sA, LDM, row0, k0, lane);
#pragma unroll
        for (int j = 0; j < 2; ++j) {
          const int n = kc * 64 + wn * 32 + j * 16;
          acc[j] = wmma_f16(a, load_b(W2, MLP_HID, n, k0, lane), acc[j]);
        }
      }
      __syncthreads();   // previous chunk's layer-3 reads of sChunk done
#pragma unroll
      for (int j = 0; j < 2; ++j) {
        const int nloc = wn * 32 + j * 16 + (lane & 15);
        const float bv = b2[kc * 64 + nloc];
        const int rb = row0 + ((lane & 16) >> 1);
#pragma unroll
        for (int r = 0; r < 8; ++r)
          sChunk[(rb + r) * LDC + nloc] = (f16_t)fmaxf(acc[j][r] + bv, 0.0f);
      }
      __syncthreads();
    }
    // layer 3 partial: acc3 += h2chunk @ W3[:, kc*64..+64)^T
#pragma unroll
    for (int kk = 0; kk < 64; kk += 32) {
      v16h a = load_a(sChunk, LDC, row0, kk, lane);
#pragma unroll
      for (int j = 0; j < 2; ++j) {
        const int n0 = wn * 32 + j * 16;
        acc3[j] = wmma_f16(a, load_b(W3, MLP_HID, n0, kc * 64 + kk, lane), acc3[j]);
      }
    }
  }

  // ---- epilogue: out = tanh(acc3 + b3) ; MAX_ACTION == 1 ----
#pragma unroll
  for (int j = 0; j < 2; ++j) {
    const int n  = wn * 32 + j * 16 + (lane & 15);
    const float bv = b3[n];
    const int rb = row0 + ((lane & 16) >> 1);
#pragma unroll
    for (int r = 0; r < 8; ++r)
      out[(wgRow + rb + r) * MLP_OUT + n] = tanhf(acc3[j][r] + bv);
  }
}

// ---------------- host launcher ----------------
extern "C" void kernel_launch(void* const* d_in, const int* in_sizes, int n_in,
                              void* d_out, int out_size, void* d_ws, size_t ws_size,
                              hipStream_t stream) {
  const float* state = (const float*)d_in[0];
  const float* WihF  = (const float*)d_in[1];
  const float* WhhF  = (const float*)d_in[2];
  const float* bih   = (const float*)d_in[3];
  const float* bhh   = (const float*)d_in[4];
  const float* W1F   = (const float*)d_in[5];
  const float* b1    = (const float*)d_in[6];
  const float* W2F   = (const float*)d_in[7];
  const float* b2    = (const float*)d_in[8];
  const float* W3F   = (const float*)d_in[9];
  const float* b3    = (const float*)d_in[10];
  float* out = (float*)d_out;

  char* ws = (char*)d_ws;
  f16_t* Wih16 = (f16_t*)(ws + OFF_WIH);
  f16_t* Whh16 = (f16_t*)(ws + OFF_WHH);
  f16_t* W1_16 = (f16_t*)(ws + OFF_W1);
  f16_t* W2_16 = (f16_t*)(ws + OFF_W2);
  f16_t* W3_16 = (f16_t*)(ws + OFF_W3);
  float* bsum  = (float*)(ws + OFF_BSUM);
  f16_t* xseq  = (f16_t*)(ws + OFF_XSEQ);
  f16_t* comb  = (f16_t*)(ws + OFF_COMB);

  const int T = 256;
  auto blk = [](int n, int t) { return (n + t - 1) / t; };

  k_cvt_f16<<<blk(2 * HDIM * HDIM, T), T, 0, stream>>>(WihF, Wih16, 2 * HDIM * HDIM);
  k_cvt_f16<<<blk(2 * HDIM * HDIM, T), T, 0, stream>>>(WhhF, Whh16, 2 * HDIM * HDIM);
  k_cvt_f16<<<blk(MLP_HID * MLP_IN, T), T, 0, stream>>>(W1F, W1_16, MLP_HID * MLP_IN);
  k_cvt_f16<<<blk(MLP_HID * MLP_HID, T), T, 0, stream>>>(W2F, W2_16, MLP_HID * MLP_HID);
  k_cvt_f16<<<blk(MLP_OUT * MLP_HID, T), T, 0, stream>>>(W3F, W3_16, MLP_OUT * MLP_HID);
  k_bias_sum<<<blk(2 * HDIM, T), T, 0, stream>>>(bih, bhh, bsum, 2 * HDIM);
  k_prep<<<blk(BATCH * HDIM, T), T, 0, stream>>>(state, xseq, comb);

  k_rnn<<<BATCH / MT, 256, 0, stream>>>(xseq, Wih16, Whh16, bsum, comb);
  k_mlp<<<BATCH / MT, 256, 0, stream>>>(comb, W1_16, b1, W2_16, b2, W3_16, b3, out);
}
